// GATLayer_68556267978813
// MI455X (gfx1250) — compile-verified, matrix operand order checked by reading, not run on hardware
//
#include <hip/hip_runtime.h>
#include <hip/hip_bf16.h>
#include <stdint.h>

#define N_NODES 8192
#define IN_DIM  256
#define OUT_DIM 64
#define HEADS   2
#define HDIM    (OUT_DIM * HEADS)   // 128
#define STRIPS  4                   // j-strips per head (occupancy / MLP)

typedef __attribute__((ext_vector_type(16))) __bf16          v16bf;
typedef __attribute__((ext_vector_type(8)))  float           v8f;
typedef __attribute__((ext_vector_type(8)))  unsigned short  u16x8;
typedef __attribute__((ext_vector_type(16))) unsigned short  u16x16;
typedef __attribute__((ext_vector_type(4)))  int             i32x4;
typedef __attribute__((ext_vector_type(4)))  float           f32x4;

__device__ __forceinline__ unsigned short f2bf(float f) {
    unsigned int u = __float_as_uint(f);
    u = (u + 0x7fffu + ((u >> 16) & 1u)) >> 16;   // round-to-nearest-even
    return (unsigned short)u;
}
__device__ __forceinline__ float bf2f(unsigned short s) {
    return __uint_as_float(((unsigned int)s) << 16);
}

// ---------------------------------------------------------------- kernel 1
__global__ void cvt_bf16_kernel(const float* __restrict__ x, const float* __restrict__ W,
                                unsigned short* __restrict__ xb, unsigned short* __restrict__ Wb) {
    int i = blockIdx.x * blockDim.x + threadIdx.x;
    int stride = gridDim.x * blockDim.x;
    const int xtotal = N_NODES * IN_DIM;
    for (int t = i; t < xtotal; t += stride) xb[t] = f2bf(x[t]);
    const int wtotal = HDIM * IN_DIM;
    for (int t = i; t < wtotal; t += stride) Wb[t] = f2bf(W[t]);
}

// ---------------------------------------------------------------- kernel 2
// h = x @ W^T, written transposed as bf16: hbT[feature][node]
__global__ __launch_bounds__(128) void proj_kernel(const unsigned short* __restrict__ xb,
                                                   const unsigned short* __restrict__ Wb,
                                                   unsigned short* __restrict__ hbT) {
    const int lane = threadIdx.x & 31;
    const int wave = threadIdx.x >> 5;
    const int hs   = lane >> 4;       // K-half select (WMMA 16-bit A/B layout)
    const int l15  = lane & 15;
    const int ibase = blockIdx.x * 16;
    const unsigned short* xrow = xb + (size_t)(ibase + l15) * IN_DIM;

    for (int t = 0; t < 2; ++t) {
        const int nbase = (wave * 2 + t) * 16;
        const unsigned short* wrow = Wb + (size_t)(nbase + l15) * IN_DIM;
        v8f acc = {};
#pragma unroll
        for (int kt = 0; kt < IN_DIM / 32; ++kt) {
            const int cb = kt * 32 + hs * 8;
            u16x8 a0 = *(const u16x8*)(xrow + cb);
            u16x8 a1 = *(const u16x8*)(xrow + cb + 16);
            u16x8 b0 = *(const u16x8*)(wrow + cb);
            u16x8 b1 = *(const u16x8*)(wrow + cb + 16);
            u16x16 av, bv;
#pragma unroll
            for (int i = 0; i < 8; ++i) { av[i] = a0[i]; av[i + 8] = a1[i];
                                          bv[i] = b0[i]; bv[i + 8] = b1[i]; }
            v16bf A = __builtin_bit_cast(v16bf, av);
            v16bf B = __builtin_bit_cast(v16bf, bv);
            acc = __builtin_amdgcn_wmma_f32_16x16x32_bf16(false, A, false, B,
                                                          (short)0, acc, false, false);
        }
        const int feat = nbase + l15;
#pragma unroll
        for (int v = 0; v < 8; ++v) {
            hbT[(size_t)feat * N_NODES + (ibase + v + hs * 8)] = f2bf(acc[v]);
        }
    }
}

// ---------------------------------------------------------------- kernel 3
__global__ void elr_kernel(const unsigned short* __restrict__ hbT,
                           const float* __restrict__ attn_l, const float* __restrict__ attn_r,
                           float* __restrict__ el, float* __restrict__ er) {
    int n = blockIdx.x * blockDim.x + threadIdx.x;
    if (n >= N_NODES) return;
#pragma unroll
    for (int h = 0; h < HEADS; ++h) {
        float sl = 0.f, sr = 0.f;
        for (int d = 0; d < OUT_DIM; ++d) {
            float hv = bf2f(hbT[(size_t)(h * OUT_DIM + d) * N_NODES + n]);
            sl += hv * attn_l[h * OUT_DIM + d];
            sr += hv * attn_r[h * OUT_DIM + d];
        }
        el[h * N_NODES + n] = sl;
        er[h * N_NODES + n] = sr;
    }
}

// ---------------------------------------------------------------- kernel 4
// Fused masked softmax + aggregation, flash style, strip-parallel.
// Block = 256 threads = 8 waves. wave w: head = w>>2, j-strip = w&3.
// Each wave keeps an independent (m, l, acc) flash state over its strip;
// states are merged through LDS (log-sum-exp merge) at the end.
__global__ __launch_bounds__(256) void gat_kernel(const int* __restrict__ adj,
                                                  const unsigned short* __restrict__ hbT,
                                                  const float* __restrict__ el,
                                                  const float* __restrict__ er,
                                                  float* __restrict__ out) {
    __shared__ float sm[HEADS * STRIPS][16];       // per-wave row max
    __shared__ float sl[HEADS][16];                // merged row sum
    __shared__ float sacc[HEADS][16][OUT_DIM];     // merged accumulator

    const int lane = threadIdx.x & 31;
    const int wid  = threadIdx.x >> 5;
    const int head  = wid >> 2;
    const int strip = wid & 3;
    const int hs   = lane >> 4;
    const int l15  = lane & 15;
    const int ibase = blockIdx.x * 16;
    const int row   = ibase + l15;

    // zero merge buffers
    for (int t = threadIdx.x; t < HEADS * 16 * OUT_DIM; t += 256)
        ((float*)sacc)[t] = 0.f;
    if (threadIdx.x < HEADS * 16) ((float*)sl)[threadIdx.x] = 0.f;
    __syncthreads();

    const int*   arow = adj + (size_t)row * N_NODES;
    const float* erh  = er + head * N_NODES;
    const float  el_v = el[head * N_NODES + row];

    const int jstart = strip * (N_NODES / STRIPS);
    const int jend   = jstart + (N_NODES / STRIPS);

    float m = -1e30f, l = 0.0f;
    v8f acc[4] = {};

    for (int jb = jstart; jb < jend; jb += 32) {
        const int c0 = jb + hs * 8;
        const int c1 = jb + 16 + hs * 8;
        // adjacency is streamed exactly once and is larger than L2 (256MB > 192MB):
        // non-temporal so it doesn't evict the reused hbT/er working set.
        i32x4 a0 = __builtin_nontemporal_load((const i32x4*)(arow + c0));
        i32x4 a1 = __builtin_nontemporal_load((const i32x4*)(arow + c0 + 4));
        i32x4 a2 = __builtin_nontemporal_load((const i32x4*)(arow + c1));
        i32x4 a3 = __builtin_nontemporal_load((const i32x4*)(arow + c1 + 4));
        f32x4 e0 = *(const f32x4*)(erh + c0);
        f32x4 e1 = *(const f32x4*)(erh + c0 + 4);
        f32x4 e2 = *(const f32x4*)(erh + c1);
        f32x4 e3 = *(const f32x4*)(erh + c1 + 4);
        if (jb + 160 < jend) __builtin_prefetch(arow + jb + 160, 0, 0);

        int   aa[16];
        float ee[16];
#pragma unroll
        for (int i = 0; i < 4; ++i) {
            aa[i] = a0[i]; aa[4 + i] = a1[i]; aa[8 + i] = a2[i]; aa[12 + i] = a3[i];
            ee[i] = e0[i]; ee[4 + i] = e1[i]; ee[8 + i] = e2[i]; ee[12 + i] = e3[i];
        }
        float ev[16];
        float tmax = -1e30f;
#pragma unroll
        for (int i = 0; i < 16; ++i) {
            float x = el_v + ee[i];
            x = (aa[i] != 0) ? x : -1e30f;
            ev[i] = x;
            tmax = fmaxf(tmax, x);
        }
        tmax = fmaxf(tmax, __shfl_xor(tmax, 16, 32));   // row pair-combine
        const float mn = fmaxf(m, tmax);
        const float alpha = __expf(m - mn);

        float ps = 0.f;
        v16bf A;
#pragma unroll
        for (int i = 0; i < 16; ++i) {
            float p = (aa[i] != 0) ? __expf(ev[i] - mn) : 0.f;
            ps += p;
            A[i] = (__bf16)p;
        }
        ps += __shfl_xor(ps, 16, 32);
        l = l * alpha + ps;
        m = mn;

        // rescale accumulators: C-row M = v + hs*8, alpha for row M in lane M
#pragma unroll
        for (int v = 0; v < 8; ++v) {
            float af = __shfl(alpha, v + hs * 8, 32);
            acc[0][v] *= af; acc[1][v] *= af; acc[2][v] *= af; acc[3][v] *= af;
        }
        // 4 feature tiles of 16: B(K=j, N=d) from transposed h (L2-resident)
#pragma unroll
        for (int nt = 0; nt < 4; ++nt) {
            const unsigned short* hp = hbT + (size_t)(head * OUT_DIM + nt * 16 + l15) * N_NODES;
            u16x8 b0 = *(const u16x8*)(hp + c0);
            u16x8 b1 = *(const u16x8*)(hp + c1);
            u16x16 bv;
#pragma unroll
            for (int i = 0; i < 8; ++i) { bv[i] = b0[i]; bv[i + 8] = b1[i]; }
            v16bf B = __builtin_bit_cast(v16bf, bv);
            acc[nt] = __builtin_amdgcn_wmma_f32_16x16x32_bf16(false, A, false, B,
                                                              (short)0, acc[nt], false, false);
        }
    }

    // ---- merge the STRIPS flash states per head (log-sum-exp) ----
    if (hs == 0) sm[wid][l15] = m;
    __syncthreads();

    const int hb = head * STRIPS;
    float M = fmaxf(fmaxf(sm[hb + 0][l15], sm[hb + 1][l15]),
                    fmaxf(sm[hb + 2][l15], sm[hb + 3][l15]));
    float f = __expf(m - M);
    if (hs == 0) atomicAdd(&sl[head][l15], l * f);

#pragma unroll
    for (int v = 0; v < 8; ++v) {
        float fr = __shfl(f, v + hs * 8, 32);
        int r = v + hs * 8;
        atomicAdd(&sacc[head][r][0 * 16 + l15], acc[0][v] * fr);
        atomicAdd(&sacc[head][r][1 * 16 + l15], acc[1][v] * fr);
        atomicAdd(&sacc[head][r][2 * 16 + l15], acc[2][v] * fr);
        atomicAdd(&sacc[head][r][3 * 16 + l15], acc[3][v] * fr);
    }
    __syncthreads();

    // normalized, coalesced, write-once (non-temporal) output
    for (int t = threadIdx.x; t < HEADS * 16 * OUT_DIM; t += 256) {
        int hh  = t / (16 * OUT_DIM);
        int rem = t - hh * 16 * OUT_DIM;
        int r   = rem / OUT_DIM;
        int d   = rem - r * OUT_DIM;
        float val = sacc[hh][r][d] / sl[hh][r];
        __builtin_nontemporal_store(val, out + (size_t)(ibase + r) * HDIM + hh * OUT_DIM + d);
    }
}

// ---------------------------------------------------------------- launch
extern "C" void kernel_launch(void* const* d_in, const int* in_sizes, int n_in,
                              void* d_out, int out_size, void* d_ws, size_t ws_size,
                              hipStream_t stream) {
    const float* x      = (const float*)d_in[0];
    const int*   adj    = (const int*)d_in[1];
    const float* W      = (const float*)d_in[2];
    const float* attn_l = (const float*)d_in[3];
    const float* attn_r = (const float*)d_in[4];
    float*       out    = (float*)d_out;

    char* ws = (char*)d_ws;
    constexpr size_t XB  = (size_t)N_NODES * IN_DIM * 2;   // 4 MB
    constexpr size_t WB  = (size_t)HDIM * IN_DIM * 2;      // 64 KB
    constexpr size_t HBT = (size_t)HDIM * N_NODES * 2;     // 2 MB
    constexpr size_t ELS = (size_t)HEADS * N_NODES * 4;    // 64 KB
    unsigned short* xb  = (unsigned short*)(ws);
    unsigned short* Wb  = (unsigned short*)(ws + XB);
    unsigned short* hbT = (unsigned short*)(ws + XB + WB);
    float*          el  = (float*)(ws + XB + WB + HBT);
    float*          er  = (float*)(ws + XB + WB + HBT + ELS);

    cvt_bf16_kernel<<<512, 256, 0, stream>>>(x, W, xb, Wb);
    proj_kernel<<<N_NODES / 16, 128, 0, stream>>>(xb, Wb, hbT);
    elr_kernel<<<(N_NODES + 255) / 256, 256, 0, stream>>>(hbT, attn_l, attn_r, el, er);
    gat_kernel<<<N_NODES / 16, 256, 0, stream>>>(adj, hbT, el, er, out);
}
